// DualStreamCrossAttention_25228637897077
// MI455X (gfx1250) — matrix operand order repaired and checked by reading, not compiled, not run on hardware
//
#include <hip/hip_runtime.h>
#include <hip/hip_bf16.h>
#include <math.h>

// Problem constants (from reference setup_inputs)
#define B_  8
#define C_  256
#define HW_ 4096
#define NH_ 8
#define HD_ 32
#define E_  64

typedef __attribute__((ext_vector_type(8)))  float        v8f;
typedef __attribute__((ext_vector_type(8)))  unsigned int v8u;
typedef __attribute__((ext_vector_type(16))) __bf16       v16bf;

__device__ __forceinline__ unsigned int f2bf(float f) {
    // round-to-nearest-even f32 -> bf16 bits
    unsigned int u = __float_as_uint(f);
    return (u + 0x7FFFu + ((u >> 16) & 1u)) >> 16;
}
__device__ __forceinline__ float sigmoid_(float x) { return 1.f / (1.f + __expf(-x)); }

// ---------------------------------------------------------------------------
// Stage A: all per-batch vector algebra that folds the attention query side.
// One workgroup of 256 threads; thread t owns output channel t in each phase.
// ---------------------------------------------------------------------------
__global__ void k_vecA(const float* __restrict__ ext_context,
                       const float* __restrict__ ext_proj_w, const float* __restrict__ ext_proj_b,
                       const float* __restrict__ ext_qkv_w,  const float* __restrict__ ext_qkv_b,
                       const float* __restrict__ road_qkv_w, const float* __restrict__ road_qkv_b,
                       const float* __restrict__ road_proj_w,const float* __restrict__ road_proj_b,
                       const float* __restrict__ road_out_w, const float* __restrict__ road_out_b,
                       const float* __restrict__ gate_road_w,const float* __restrict__ gate_road_b,
                       float* __restrict__ ext_p, float* __restrict__ q, float* __restrict__ extv,
                       float* __restrict__ u, float* __restrict__ u2, float* __restrict__ kconst,
                       float* __restrict__ road_vec, float* __restrict__ rg_const) {
    const int t = threadIdx.x;
    const float scale = 0.17677669529663687f; // 32^-0.5

    // 1) ext_p = ext_proj_w @ ext_context + b
    for (int b = 0; b < B_; ++b) {
        float a = ext_proj_b[t];
        for (int e = 0; e < E_; ++e) a += ext_proj_w[t * E_ + e] * ext_context[b * E_ + e];
        ext_p[b * C_ + t] = a;
    }
    __syncthreads();
    // 2) eqkv rows t (q) and 512+t (v)
    for (int b = 0; b < B_; ++b) {
        float aq = ext_qkv_b[t], av_ = ext_qkv_b[2 * C_ + t];
        for (int c = 0; c < C_; ++c) {
            float x = ext_p[b * C_ + c];
            aq  += ext_qkv_w[t * C_ + c] * x;
            av_ += ext_qkv_w[(2 * C_ + t) * C_ + c] * x;
        }
        q[b * C_ + t] = aq;
        extv[b * C_ + t] = av_;
    }
    __syncthreads();
    // 3) u(b,n,c=t) = sum_d q(b,n,d) * Wk[(n,d) row], Wk row = 3*(n*32+d)+1 of road_qkv_w
    for (int b = 0; b < B_; ++b)
        for (int n = 0; n < NH_; ++n) {
            float a = 0.f;
            for (int d = 0; d < HD_; ++d)
                a += q[b * C_ + n * HD_ + d] * road_qkv_w[(size_t)(3 * (n * HD_ + d) + 1) * C_ + t];
            u[(b * NH_ + n) * C_ + t] = a;
        }
    __syncthreads();
    // 4) u2 = (u @ road_proj_w) * scale ; kconst = (q.bk + u.bp) * scale
    for (int b = 0; b < B_; ++b)
        for (int n = 0; n < NH_; ++n) {
            float a = 0.f;
            for (int c = 0; c < C_; ++c)
                a += u[(b * NH_ + n) * C_ + c] * road_proj_w[(size_t)c * C_ + t];
            u2[(b * NH_ + n) * C_ + t] = a * scale;
        }
    if (t < B_ * NH_) {
        int b = t >> 3, n = t & 7;
        float a = 0.f;
        for (int d = 0; d < HD_; ++d) a += q[b * C_ + n * HD_ + d] * road_qkv_b[3 * (n * HD_ + d) + 1];
        for (int c = 0; c < C_; ++c)  a += u[(b * NH_ + n) * C_ + c] * road_proj_b[c];
        kconst[t] = a * scale;
    }
    // 5) road_vec = road_out_w @ extv + b
    for (int b = 0; b < B_; ++b) {
        float a = road_out_b[t];
        for (int c = 0; c < C_; ++c) a += road_out_w[t * C_ + c] * extv[b * C_ + c];
        road_vec[b * C_ + t] = a;
    }
    __syncthreads();
    // 6) rg_const = gate_road_w[:,256:512] @ road_vec + gate_road_b
    for (int b = 0; b < B_; ++b) {
        float a = gate_road_b[t];
        for (int c = 0; c < C_; ++c) a += gate_road_w[t * 2 * C_ + C_ + c] * road_vec[b * C_ + c];
        rg_const[b * C_ + t] = a;
    }
}

// ---------------------------------------------------------------------------
// Pack gate_road_w[:, :256] into bf16 WMMA A-fragment order (ISA 7.12.2).
// wfrag[((oct*8 + kt)*32 + lane)*8 + v] = packed {K=k0, K=k0+1}.
// ---------------------------------------------------------------------------
__global__ void k_wfrag(const float* __restrict__ gw, unsigned int* __restrict__ wfrag) {
    int idx  = blockIdx.x * 256 + threadIdx.x;     // 32768 total
    int v    = idx & 7;
    int lane = (idx >> 3) & 31;
    int kt   = (idx >> 8) & 7;
    int oct  = idx >> 11;                          // 0..15
    int row  = oct * 16 + (lane & 15);
    int kb   = (lane >> 4) * 8;
    int k0   = (v < 4) ? (2 * v + kb) : (16 + 2 * (v - 4) + kb);
    int c    = kt * 32 + k0;
    unsigned lo = f2bf(gw[(size_t)row * 2 * C_ + c]);
    unsigned hi = f2bf(gw[(size_t)row * 2 * C_ + c + 1]);
    wfrag[idx] = lo | (hi << 16);
}

// ---------------------------------------------------------------------------
// scores(b,n,p) = u2(b,n,:) . road_feat(b,:,p) + kconst   (already *scale)
// block = (b, 256-wide p chunk); one pass over road_feat.
// ---------------------------------------------------------------------------
__global__ void k_scores(const float* __restrict__ road, const float* __restrict__ u2,
                         const float* __restrict__ kconst, float* __restrict__ scores) {
    int b = blockIdx.x >> 4;
    int p = (blockIdx.x & 15) * 256 + threadIdx.x;
    __shared__ float su[NH_][C_];
    for (int j = 0; j < NH_; ++j) su[j][threadIdx.x] = u2[(b * NH_ + j) * C_ + threadIdx.x];
    __syncthreads();
    float acc[NH_];
#pragma unroll
    for (int n = 0; n < NH_; ++n) acc[n] = kconst[b * NH_ + n];
    const float* rb = road + (size_t)b * C_ * HW_ + p;
    for (int c = 0; c < C_; ++c) {
        float x = rb[(size_t)c * HW_];
#pragma unroll
        for (int n = 0; n < NH_; ++n) acc[n] += su[n][c] * x;
    }
#pragma unroll
    for (int n = 0; n < NH_; ++n) scores[((size_t)(b * NH_ + n)) * HW_ + p] = acc[n];
}

// softmax over HW per (b,n) row, in place
__global__ void k_softmax(float* __restrict__ s_) {
    float* s = s_ + (size_t)blockIdx.x * HW_;
    int t = threadIdx.x;
    __shared__ float red[256];
    float m = -3.4e38f;
    for (int p = t; p < HW_; p += 256) m = fmaxf(m, s[p]);
    red[t] = m; __syncthreads();
    for (int st = 128; st > 0; st >>= 1) { if (t < st) red[t] = fmaxf(red[t], red[t + st]); __syncthreads(); }
    m = red[0]; __syncthreads();
    float sum = 0.f;
    for (int p = t; p < HW_; p += 256) { float e = __expf(s[p] - m); s[p] = e; sum += e; }
    red[t] = sum; __syncthreads();
    for (int st = 128; st > 0; st >>= 1) { if (t < st) red[t] += red[t + st]; __syncthreads(); }
    float inv = 1.f / red[0];
    for (int p = t; p < HW_; p += 256) s[p] *= inv;
}

// av(b,n,c) = sum_p attn(b,n,p) * road_feat(b,c,p) ; block = (b,c)
__global__ void k_av(const float* __restrict__ road, const float* __restrict__ attn,
                     float* __restrict__ av) {
    int b = blockIdx.x >> 8, c = blockIdx.x & 255;
    const float* rb = road + ((size_t)(b * C_ + c)) * HW_;
    const float* ab = attn + (size_t)b * NH_ * HW_;
    int t = threadIdx.x;
    float acc[NH_];
#pragma unroll
    for (int n = 0; n < NH_; ++n) acc[n] = 0.f;
    for (int p = t; p < HW_; p += 256) {
        float x = rb[p];
#pragma unroll
        for (int n = 0; n < NH_; ++n) acc[n] += ab[(size_t)n * HW_ + p] * x;
    }
    __shared__ float red[NH_][256];
#pragma unroll
    for (int n = 0; n < NH_; ++n) red[n][t] = acc[n];
    __syncthreads();
    for (int st = 128; st > 0; st >>= 1) {
        if (t < st) {
#pragma unroll
            for (int n = 0; n < NH_; ++n) red[n][t] += red[n][t + st];
        }
        __syncthreads();
    }
    if (t < NH_) av[(size_t)(b * NH_ + t) * C_ + c] = red[t][0];
}

// lfmean(b,c) = mean over HW of low_freq ; block = (b,c)
__global__ void k_lfmean(const float* __restrict__ low, float* __restrict__ lfmean) {
    int bc = blockIdx.x;
    const float* lb = low + (size_t)bc * HW_;
    int t = threadIdx.x;
    float acc = 0.f;
    for (int p = t; p < HW_; p += 256) acc += lb[p];
    __shared__ float red[256];
    red[t] = acc; __syncthreads();
    for (int st = 128; st > 0; st >>= 1) { if (t < st) red[t] += red[t + st]; __syncthreads(); }
    if (t == 0) lfmean[bc] = red[0] * (1.f / (float)HW_);
}

// ---------------------------------------------------------------------------
// Stage B vector algebra: value-side fold, ext_vec, ext gate.
// ---------------------------------------------------------------------------
__global__ void k_vecB(const float* __restrict__ av,
                       const float* __restrict__ road_proj_w, const float* __restrict__ road_proj_b,
                       const float* __restrict__ road_qkv_w,  const float* __restrict__ road_qkv_b,
                       const float* __restrict__ ext_out_w,   const float* __restrict__ ext_out_b,
                       const float* __restrict__ gate_ext_w,  const float* __restrict__ gate_ext_b,
                       const float* __restrict__ lfmean,
                       float* __restrict__ avp, float* __restrict__ ext_head,
                       float* __restrict__ ext_vec, float* __restrict__ ge) {
    const int t = threadIdx.x;
    // avp(b,n,c=t) = road_proj_w[t,:] . av(b,n,:) + bp[t]
    for (int b = 0; b < B_; ++b)
        for (int n = 0; n < NH_; ++n) {
            float a = road_proj_b[t];
            for (int c = 0; c < C_; ++c) a += road_proj_w[(size_t)t * C_ + c] * av[(b * NH_ + n) * C_ + c];
            avp[(b * NH_ + n) * C_ + t] = a;
        }
    __syncthreads();
    // ext_head(b, t=n*32+d) = Wv[(n,d) row=3t+2] . avp(b,n,:) + bv
    for (int b = 0; b < B_; ++b) {
        int n = t >> 5;
        float a = road_qkv_b[3 * t + 2];
        for (int c = 0; c < C_; ++c) a += road_qkv_w[(size_t)(3 * t + 2) * C_ + c] * avp[(b * NH_ + n) * C_ + c];
        ext_head[b * C_ + t] = a;
    }
    __syncthreads();
    // ext_vec = ext_out_w @ ext_head + b
    for (int b = 0; b < B_; ++b) {
        float a = ext_out_b[t];
        for (int c = 0; c < C_; ++c) a += ext_out_w[t * C_ + c] * ext_head[b * C_ + c];
        ext_vec[b * C_ + t] = a;
    }
    __syncthreads();
    // ge = sigmoid(Wg[:, :256]@lfmean + Wg[:, 256:]@ext_vec + b)
    for (int b = 0; b < B_; ++b) {
        float a = gate_ext_b[t];
        for (int c = 0; c < C_; ++c)
            a += gate_ext_w[t * 2 * C_ + c] * lfmean[b * C_ + c]
               + gate_ext_w[t * 2 * C_ + C_ + c] * ext_vec[b * C_ + c];
        ge[b * C_ + t] = sigmoid_(a);
    }
}

// ext_final = low*(1-g) + ext_vec*g (elementwise, g per (b,c))
__global__ void k_extfinal(const float* __restrict__ low, const float* __restrict__ ext_vec,
                           const float* __restrict__ ge, float* __restrict__ out) {
    size_t i = (size_t)blockIdx.x * 256 + threadIdx.x;
    size_t bc = i / HW_;
    float g = ge[bc];
    out[i] = low[i] * (1.f - g) + ext_vec[bc] * g;
}

// ---------------------------------------------------------------------------
// Road-gate conv via bf16 WMMA:
//   glin(b,oc,p) = Wg1 @ high_freq + rg_const ;  out = high*(1-sig) + road_vec*sig
// Block: 256 thr (8 waves), computes all 256 oc for a 64-wide p slab of batch b.
// Wave w owns oc-tiles {2w,2w+1} x p-tiles {0..3}; K=256 in 8 steps of 32,
// X slab staged through LDS in B-fragment order (packed bf16 pairs).
// ---------------------------------------------------------------------------
__global__ void k_roadgate(const float* __restrict__ high, const unsigned int* __restrict__ wfrag,
                           const float* __restrict__ rg_const, const float* __restrict__ road_vec,
                           float* __restrict__ out_road) {
    const int b     = blockIdx.x >> 6;
    const int pbase = (blockIdx.x & 63) * 64;
    const int tid   = threadIdx.x;
    const int wave  = tid >> 5, lane = tid & 31;
    const float* hb = high + (size_t)b * C_ * HW_;

    __shared__ unsigned int xfrag[4][32][8]; // 4 p-tiles, B-fragment order, 4 KB

    v8f acc[2][4];
#pragma unroll
    for (int i = 0; i < 2; ++i)
#pragma unroll
        for (int j = 0; j < 4; ++j) { v8f z = {0.f,0.f,0.f,0.f,0.f,0.f,0.f,0.f}; acc[i][j] = z; }

    for (int kt = 0; kt < 8; ++kt) {
        __syncthreads(); // previous iteration's fragments consumed
        // stage 32 channels x 64 p of high_freq -> bf16 pairs in B-frag order
#pragma unroll
        for (int j = 0; j < 4; ++j) {
            int idx = tid + j * 256;          // 0..1023
            int p   = idx & 63;
            int cp  = idx >> 6;               // channel pair 0..15
            int c   = kt * 32 + cp * 2;
            float x0 = hb[(size_t)c * HW_ + pbase + p];
            float x1 = hb[(size_t)(c + 1) * HW_ + pbase + p];
            int pt = p >> 4, n = p & 15;
            int lg = (cp >> 2) & 1;           // ((local c)%16)/8
            int vv = (cp >> 3) * 4 + (cp & 3); // (local c)/16*4 + ((local c)%8)/2
            xfrag[pt][lg * 16 + n][vv] = f2bf(x0) | (f2bf(x1) << 16);
        }
        __syncthreads();

        v16bf afr[2];
#pragma unroll
        for (int oi = 0; oi < 2; ++oi) {
            const v8u* ap = (const v8u*)(wfrag + ((size_t)((wave * 2 + oi) * 8 + kt) * 32 + lane) * 8);
            afr[oi] = __builtin_bit_cast(v16bf, *ap);
        }
#pragma unroll
        for (int pi = 0; pi < 4; ++pi) {
            v8u braw = *(const v8u*)&xfrag[pi][lane][0];
            v16bf bfr = __builtin_bit_cast(v16bf, braw);
#pragma unroll
            for (int oi = 0; oi < 2; ++oi)
                acc[oi][pi] = __builtin_amdgcn_wmma_f32_16x16x32_bf16(
                    false, afr[oi], false, bfr, (short)0, acc[oi][pi], false, false);
        }
    }

    // epilogue: D(m,n) in lane = n + 16*(m/8), VGPR v = m%8 (ISA 7.12.2)
    const int grp = lane >> 4, n = lane & 15;
#pragma unroll
    for (int oi = 0; oi < 2; ++oi)
#pragma unroll
        for (int pi = 0; pi < 4; ++pi)
#pragma unroll
            for (int v = 0; v < 8; ++v) {
                int oc = (wave * 2 + oi) * 16 + v + 8 * grp;
                int p  = pbase + pi * 16 + n;
                float g = sigmoid_(acc[oi][pi][v] + rg_const[b * C_ + oc]);
                float h = hb[(size_t)oc * HW_ + p];
                out_road[((size_t)(b * C_ + oc)) * HW_ + p] =
                    h * (1.f - g) + road_vec[b * C_ + oc] * g;
            }
}

// ---------------------------------------------------------------------------
extern "C" void kernel_launch(void* const* d_in, const int* in_sizes, int n_in,
                              void* d_out, int out_size, void* d_ws, size_t ws_size,
                              hipStream_t stream) {
    (void)in_sizes; (void)n_in; (void)out_size; (void)ws_size;
    const float* low        = (const float*)d_in[0];
    const float* high       = (const float*)d_in[1];
    const float* ext_ctx    = (const float*)d_in[2];
    const float* road       = (const float*)d_in[3];
    const float* ext_proj_w = (const float*)d_in[4];
    const float* ext_proj_b = (const float*)d_in[5];
    const float* road_proj_w= (const float*)d_in[6];
    const float* road_proj_b= (const float*)d_in[7];
    const float* ext_qkv_w  = (const float*)d_in[8];
    const float* ext_qkv_b  = (const float*)d_in[9];
    const float* road_qkv_w = (const float*)d_in[10];
    const float* road_qkv_b = (const float*)d_in[11];
    const float* ext_out_w  = (const float*)d_in[12];
    const float* ext_out_b  = (const float*)d_in[13];
    const float* road_out_w = (const float*)d_in[14];
    const float* road_out_b = (const float*)d_in[15];
    const float* gate_ext_w = (const float*)d_in[16];
    const float* gate_ext_b = (const float*)d_in[17];
    const float* gate_road_w= (const float*)d_in[18];
    const float* gate_road_b= (const float*)d_in[19];

    float* ws = (float*)d_ws;
    // workspace layout (float slots)
    float* w_ext_p    = ws + 0;        // 2048
    float* w_q        = ws + 2048;     // 2048
    float* w_extv     = ws + 4096;     // 2048
    float* w_u        = ws + 6144;     // 16384
    float* w_u2       = ws + 22528;    // 16384
    float* w_kconst   = ws + 38912;    // 64
    float* w_road_vec = ws + 38976;    // 2048
    float* w_rg_const = ws + 41024;    // 2048
    float* w_scores   = ws + 43072;    // 262144 (becomes attn in place)
    float* w_av       = ws + 305216;   // 16384
    float* w_avp      = ws + 321600;   // 16384
    float* w_ext_head = ws + 337984;   // 2048
    float* w_ext_vec  = ws + 340032;   // 2048
    float* w_ge       = ws + 342080;   // 2048
    float* w_lfmean   = ws + 344128;   // 2048
    unsigned int* w_wfrag = (unsigned int*)(ws + 346176); // 32768 u32 (32B aligned)

    float* out_ext  = (float*)d_out;                    // (B,C,H,W)
    float* out_road = (float*)d_out + (size_t)B_ * C_ * HW_;

    k_vecA<<<1, 256, 0, stream>>>(ext_ctx, ext_proj_w, ext_proj_b, ext_qkv_w, ext_qkv_b,
                                  road_qkv_w, road_qkv_b, road_proj_w, road_proj_b,
                                  road_out_w, road_out_b, gate_road_w, gate_road_b,
                                  w_ext_p, w_q, w_extv, w_u, w_u2, w_kconst,
                                  w_road_vec, w_rg_const);
    k_wfrag<<<128, 256, 0, stream>>>(gate_road_w, w_wfrag);
    k_scores<<<B_ * 16, 256, 0, stream>>>(road, w_u2, w_kconst, w_scores);
    k_softmax<<<B_ * NH_, 256, 0, stream>>>(w_scores);
    k_av<<<B_ * C_, 256, 0, stream>>>(road, w_scores, w_av);
    k_lfmean<<<B_ * C_, 256, 0, stream>>>(low, w_lfmean);
    k_vecB<<<1, 256, 0, stream>>>(w_av, road_proj_w, road_proj_b, road_qkv_w, road_qkv_b,
                                  ext_out_w, ext_out_b, gate_ext_w, gate_ext_b, w_lfmean,
                                  w_avp, w_ext_head, w_ext_vec, w_ge);
    k_extfinal<<<(B_ * C_ * HW_) / 256, 256, 0, stream>>>(low, w_ext_vec, w_ge, out_ext);
    k_roadgate<<<B_ * 64, 256, 0, stream>>>(high, w_wfrag, w_rg_const, w_road_vec, out_road);
}
